// GatedLinearAttention_558345749042
// MI455X (gfx1250) — matrix operand order repaired and checked by reading, not compiled, not run on hardware
//
#include <hip/hip_runtime.h>
#include <hip/hip_bf16.h>

// GLA forward for MI455X (gfx1250, wave32, WMMA bf16, async LDS staging).
// B=2, N=1024, D=1024, H=4, DK=128, DV=256, LR=16, chunk=16.

typedef __attribute__((ext_vector_type(16))) __bf16 v16bf;
typedef __attribute__((ext_vector_type(8)))  float  v8f;

union FragB { unsigned u[8]; v16bf v; };
union AccF  { v8f v; float f[8]; };

__device__ __forceinline__ unsigned f2bf1(float f) {
    unsigned u = __float_as_uint(f);
    u += 0x7fffu + ((u >> 16) & 1u);          // round-to-nearest-even
    return u >> 16;
}
__device__ __forceinline__ unsigned pack2(float lo, float hi) {
    return f2bf1(lo) | (f2bf1(hi) << 16);
}
__device__ __forceinline__ v8f wmma_bf16(const FragB& a, const FragB& b, v8f c) {
    return __builtin_amdgcn_wmma_f32_16x16x32_bf16(
        false, a.v, false, b.v, (short)0, c, false, false);
}

#if defined(__has_builtin)
#if __has_builtin(__builtin_amdgcn_global_load_async_to_lds_b128)
#define HAVE_ASYNC_LDS 1
#endif
#endif

#ifdef HAVE_ASYNC_LDS
#define AS1 __attribute__((address_space(1)))
#define AS3 __attribute__((address_space(3)))
typedef int async_v4i __attribute__((__vector_size__(4 * sizeof(int))));
__device__ __forceinline__ void async_cp16(const void* g, void* l) {
    __builtin_amdgcn_global_load_async_to_lds_b128(
        (AS1 async_v4i*)g, (AS3 async_v4i*)l, 0, 0);
}
#endif

// ---- Fragment loaders (contiguous k = 16*half + s order, wave32) -------------------------
// All bf16 operands are staged as u32 = (elem[2k2], elem[2k2+1]) pairs along K,
// so each fragment is 8 coalescible 32-bit DS reads.
// A 16x32: lane-> M = L%16, half = L>>4; VGPR g -> k2 local:
//   k2 = 8*half + g
__device__ __forceinline__ FragB loadA_u32(const unsigned* p, int stride2,
                                           int mbase, int k2base, int k2valid,
                                           int l16, int half) {
    FragB f;
    int m = mbase + l16;
#pragma unroll
    for (int g = 0; g < 8; ++g) {
        int k2 = 8 * half + g;
        f.u[g] = (k2 < k2valid) ? p[m * stride2 + k2base + k2] : 0u;
    }
    return f;
}
// B 32x16: lane-> N = L%16, half = L>>4; VGPR g -> k2 = 8*half + g
__device__ __forceinline__ FragB loadB_u32(const unsigned* p, int stride,
                                           int k2base, int nbase, int k2valid,
                                           int l16, int half) {
    FragB f;
    int n = nbase + l16;
#pragma unroll
    for (int g = 0; g < 8; ++g) {
        int k2 = 8 * half + g;
        f.u[g] = (k2 < k2valid) ? p[(k2base + k2) * stride + n] : 0u;
    }
    return f;
}
// B fragment from fp32 LDS matrix (state S), converting to bf16 on the fly
__device__ __forceinline__ FragB loadB_f32(const float* p, int stride,
                                           int kbase, int nbase, int l16, int half) {
    FragB f;
    int n = nbase + l16;
#pragma unroll
    for (int g = 0; g < 8; ++g) {
        int k = kbase + 16 * half + 2 * g;
        f.u[g] = pack2(p[k * stride + n], p[(k + 1) * stride + n]);
    }
    return f;
}

// ---------------------------------------------------------------------------
// bf16 WMMA GEMM: C[M,N](fp32) = A[M,K](bf16 row-major) * Bpk[K/2,N](u32 pairs)
// 128x128 tile / 256 threads (8 waves, 4x2 tiles each), K-step 32.
// ---------------------------------------------------------------------------
__global__ __launch_bounds__(256) void gemm_bf16_kernel(
    const unsigned short* __restrict__ A, const unsigned* __restrict__ Bpk,
    float* __restrict__ C, int M, int N, int K) {
    __shared__ __align__(16) unsigned As[128][16];   // [m][k2]
    __shared__ __align__(16) unsigned Bs[16][128];   // [k2][n]

    const int tid  = threadIdx.x;
    const int lane = tid & 31, wave = tid >> 5;
    const int half = lane >> 4, l16 = lane & 15;
    const int m0 = blockIdx.y * 128, n0 = blockIdx.x * 128;
    const int wr = wave >> 2, wc = wave & 3;
    const int K2 = K >> 1;

    v8f acc[4][2];
#pragma unroll
    for (int a = 0; a < 4; ++a)
#pragma unroll
        for (int b = 0; b < 2; ++b) acc[a][b] = v8f{};

    for (int kt = 0; kt < K; kt += 32) {
        const int kt2 = kt >> 1;
#ifdef HAVE_ASYNC_LDS
        // ASYNCcnt-tracked byte copies: A tile 8KB, B tile 8KB, 16B per lane op
#pragma unroll
        for (int i = 0; i < 2; ++i) {
            int p = tid + i * 256;          // 0..511
            int m = p >> 2, c = p & 3;      // 4x16B chunks per 64B row
            async_cp16(A + (size_t)(m0 + m) * K + kt + c * 8, &As[m][c * 4]);
        }
#pragma unroll
        for (int i = 0; i < 2; ++i) {
            int p = tid + i * 256;
            int k2 = p >> 5, c = p & 31;    // 32x16B chunks per 512B row
            async_cp16(Bpk + (size_t)(kt2 + k2) * N + n0 + c * 4, &Bs[k2][c * 4]);
        }
        asm volatile("s_wait_asynccnt 0" ::: "memory");
#else
        const unsigned* A32 = (const unsigned*)A;
#pragma unroll
        for (int i = 0; i < 8; ++i) {
            int p = tid * 8 + i;
            int m = p >> 4, k2 = p & 15;
            As[m][k2] = A32[(size_t)(m0 + m) * K2 + kt2 + k2];
        }
#pragma unroll
        for (int i = 0; i < 8; ++i) {
            int p = tid * 8 + i;
            int k2 = p >> 7, n = p & 127;
            Bs[k2][n] = Bpk[(size_t)(kt2 + k2) * N + n0 + n];
        }
#endif
        __syncthreads();

        FragB af[4], bf[2];
#pragma unroll
        for (int a = 0; a < 4; ++a)
            af[a] = loadA_u32(&As[0][0], 16, (wr * 4 + a) * 16, 0, 16, l16, half);
#pragma unroll
        for (int b = 0; b < 2; ++b)
            bf[b] = loadB_u32(&Bs[0][0], 128, 0, (wc * 2 + b) * 16, 16, l16, half);
#pragma unroll
        for (int a = 0; a < 4; ++a)
#pragma unroll
            for (int b = 0; b < 2; ++b)
                acc[a][b] = wmma_bf16(af[a], bf[b], acc[a][b]);
        __syncthreads();
    }

#pragma unroll
    for (int a = 0; a < 4; ++a)
#pragma unroll
        for (int b = 0; b < 2; ++b) {
            AccF r; r.v = acc[a][b];
            int col = n0 + (wc * 2 + b) * 16 + l16;
#pragma unroll
            for (int rr = 0; rr < 8; ++rr) {
                int row = m0 + (wr * 4 + a) * 16 + rr + 8 * half;
                C[(size_t)row * N + col] = r.f[rr];
            }
        }
}

// ---------------------------------------------------------------------------
// fp32 -> bf16 elementwise (A-operand layout, row-major)
__global__ __launch_bounds__(256) void cvt_bf16_kernel(
    const float* __restrict__ in, unsigned short* __restrict__ out, int n) {
    int gid = blockIdx.x * 256 + threadIdx.x;
    if (gid < n) out[gid] = (unsigned short)f2bf1(in[gid]);
}
// fp32 [K][N] -> packed u32 [K/2][N] (B-operand layout: pairs along K)
__global__ __launch_bounds__(256) void pack_weight_kernel(
    const float* __restrict__ W, unsigned* __restrict__ out, int K, int N) {
    int gid = blockIdx.x * 256 + threadIdx.x;
    int total = (K >> 1) * N;
    if (gid >= total) return;
    int k2 = gid / N, n = gid - k2 * N;
    out[gid] = pack2(W[(size_t)(2 * k2) * N + n], W[(size_t)(2 * k2 + 1) * N + n]);
}

// ---------------------------------------------------------------------------
// Low-rank gate path
// ---------------------------------------------------------------------------
__global__ __launch_bounds__(256) void glr_kernel(
    const float* __restrict__ x, const float* __restrict__ W1,
    float* __restrict__ glr) {
    int gid = blockIdx.x * 256 + threadIdx.x;   // 2048*16
    if (gid >= 2048 * 16) return;
    int row = gid >> 4, j = gid & 15;
    const float* xr = x + (size_t)row * 1024;
    float acc = 0.f;
    for (int k = 0; k < 1024; ++k) acc += xr[k] * W1[k * 16 + j];
    glr[gid] = acc;
}
// g = clamp(log_sigmoid(glr@Wgk2 + b)/16, -3)
__global__ __launch_bounds__(256) void gate_decay_kernel(
    const float* __restrict__ glr, const float* __restrict__ W2,
    const float* __restrict__ bias, float* __restrict__ g) {
    int gid = blockIdx.x * 256 + threadIdx.x;   // 2048*512
    if (gid >= 2048 * 512) return;
    int row = gid >> 9, c = gid & 511;
    const float* gr = glr + (size_t)row * 16;
    float acc = bias[c];
#pragma unroll
    for (int l = 0; l < 16; ++l) acc += gr[l] * W2[l * 512 + c];
    float ls = fminf(acc, 0.f) - log1pf(__expf(-fabsf(acc)));
    g[gid] = fmaxf(ls * (1.f / 16.f), -3.f);
}

// ---------------------------------------------------------------------------
// Chunked GLA scan: one workgroup per (b,h); state S[128][256] fp32 in LDS.
// chunk=16:  A = (Q e^G)(K e^-G)^T (masked);  O = A V + (Q e^G) S
//            S = diag(e^Gc) S + (K e^{Gc-G})^T V
// ---------------------------------------------------------------------------
__global__ __launch_bounds__(256) void gla_scan_kernel(
    const float* __restrict__ q, const float* __restrict__ k,
    const float* __restrict__ v, const float* __restrict__ g,
    float* __restrict__ o) {
    extern __shared__ char smem[];
    float*          S    = (float*)smem;                   // [128][256] fp32
    unsigned*       Qs2  = (unsigned*)(smem + 131072);     // [16][64]  A: q*e^G, pairs along dk
    unsigned*       Kng2 = Qs2  + 16 * 64;                 // [64][16]  B: k*e^-G, pairs along dk
    unsigned*       Kc2  = Kng2 + 64 * 16;                 // [128][8]  A: k*e^{Gc-G}, pairs along t
    unsigned*       VB2  = Kc2  + 128 * 8;                 // [8][256]  B: v, pairs along t
    unsigned short* Ab   = (unsigned short*)(VB2 + 8 * 256); // [16][16] masked A
    float*          eC   = (float*)(Ab + 256);             // [128] e^{Gc}

    const int tid  = threadIdx.x;
    const int lane = tid & 31, wave = tid >> 5;
    const int half = lane >> 4, l16 = lane & 15;
    const int b = blockIdx.x >> 2, h = blockIdx.x & 3;

    const float* qb = q + (size_t)b * 1024 * 512 + h * 128;
    const float* kb = k + (size_t)b * 1024 * 512 + h * 128;
    const float* gb = g + (size_t)b * 1024 * 512 + h * 128;
    const float* vb = v + (size_t)b * 1024 * 1024 + h * 256;
    float*       ob = o + (size_t)b * 1024 * 1024 + h * 256;

    for (int i = tid; i < 128 * 256; i += 256) S[i] = 0.f;

    for (int c = 0; c < 64; ++c) {
        __syncthreads();
        const int t0 = c * 16;

        // ---- phase 1: decays + packed bf16 staging (thread owns dk pair) ---
        if (tid < 64) {
            const int dk2 = tid, dk0 = 2 * tid;
            float Gc0 = 0.f, Gc1 = 0.f;
            for (int t = 0; t < 16; ++t) {
                size_t ro = (size_t)(t0 + t) * 512 + dk0;
                Gc0 += gb[ro]; Gc1 += gb[ro + 1];
            }
            eC[dk0] = __expf(Gc0); eC[dk0 + 1] = __expf(Gc1);
            float G0 = 0.f, G1 = 0.f, pc0 = 0.f, pc1 = 0.f;
            for (int t = 0; t < 16; ++t) {
                size_t ro = (size_t)(t0 + t) * 512 + dk0;
                G0 += gb[ro]; G1 += gb[ro + 1];
                float e0 = __expf(G0), e1 = __expf(G1);
                Qs2[t * 64 + dk2]  = pack2(qb[ro] * e0, qb[ro + 1] * e1);
                Kng2[dk2 * 16 + t] = pack2(kb[ro] * __expf(-G0), kb[ro + 1] * __expf(-G1));
                float c0 = kb[ro] * __expf(Gc0 - G0), c1 = kb[ro + 1] * __expf(Gc1 - G1);
                if (t & 1) {
                    Kc2[dk0 * 8 + (t >> 1)]       = pack2(pc0, c0);
                    Kc2[(dk0 + 1) * 8 + (t >> 1)] = pack2(pc1, c1);
                } else { pc0 = c0; pc1 = c1; }
            }
        }
        for (int i = tid; i < 8 * 256; i += 256) {
            int t2 = i >> 8, dv = i & 255;
            VB2[i] = pack2(vb[(size_t)(t0 + 2 * t2) * 1024 + dv],
                           vb[(size_t)(t0 + 2 * t2 + 1) * 1024 + dv]);
        }
        __syncthreads();

        // ---- phase 2: wave 0 computes masked intra-chunk A [16x16] ---------
        if (wave == 0) {
            v8f a = v8f{};
#pragma unroll
            for (int kk = 0; kk < 4; ++kk) {
                FragB af = loadA_u32(Qs2, 64, 0, kk * 16, 16, l16, half);
                FragB bf = loadB_u32(Kng2, 16, kk * 16, 0, 16, l16, half);
                a = wmma_bf16(af, bf, a);
            }
            AccF r; r.v = a;
            int s = l16;
#pragma unroll
            for (int rr = 0; rr < 8; ++rr) {
                int t = rr + 8 * half;
                Ab[t * 16 + s] = (unsigned short)f2bf1((s <= t) ? r.f[rr] : 0.f);
            }
        }
        __syncthreads();

        // ---- phase 3: outputs + state update; wave owns dv-tiles {w, w+8} --
#pragma unroll
        for (int rep = 0; rep < 2; ++rep) {
            const int nb = (wave + rep * 8) * 16;
            FragB vfrag = loadB_u32(VB2, 256, 0, nb, 8, l16, half);  // shared by O & S
            // O = A@V + (Q e^G)@S
            v8f oacc = v8f{};
            {
                FragB af = loadA_u32((const unsigned*)Ab, 8, 0, 0, 8, l16, half);
                oacc = wmma_bf16(af, vfrag, oacc);
            }
#pragma unroll
            for (int kk = 0; kk < 4; ++kk) {
                FragB af = loadA_u32(Qs2, 64, 0, kk * 16, 16, l16, half);
                FragB bf = loadB_f32(S, 256, kk * 32, nb, l16, half);
                oacc = wmma_bf16(af, bf, oacc);
            }
            AccF r; r.v = oacc;
            int col = nb + l16;
#pragma unroll
            for (int rr = 0; rr < 8; ++rr) {
                int t = rr + 8 * half;
                ob[(size_t)(t0 + t) * 1024 + col] = r.f[rr];
            }
            // S = e^{Gc}*S + Kc^T @ V   (per dk-tile)
#pragma unroll
            for (int i8 = 0; i8 < 8; ++i8) {
                v8f sacc = v8f{};
                FragB af = loadA_u32(Kc2, 8, i8 * 16, 0, 8, l16, half);
                sacc = wmma_bf16(af, vfrag, sacc);
                AccF sr; sr.v = sacc;
#pragma unroll
                for (int rr = 0; rr < 8; ++rr) {
                    int dk = i8 * 16 + rr + 8 * half;
                    int idx = dk * 256 + col;
                    S[idx] = eC[dk] * S[idx] + sr.f[rr];
                }
            }
        }
    }
}

// ---------------------------------------------------------------------------
// y(bf16) = RMSNorm_head(o) * rms_w * silu(gate)
// ---------------------------------------------------------------------------
__global__ __launch_bounds__(256) void rms_silu_kernel(
    const float* __restrict__ o, const float* __restrict__ gate,
    const float* __restrict__ rms_w, unsigned short* __restrict__ yb) {
    __shared__ float red[256];
    __shared__ float hs[4];
    int row = blockIdx.x, tid = threadIdx.x;
    int h = tid >> 6, l = tid & 63;
    const float* orow = o + (size_t)row * 1024;
    float vals[4], ss = 0.f;
#pragma unroll
    for (int i = 0; i < 4; ++i) {
        float x_ = orow[h * 256 + l * 4 + i];
        vals[i] = x_;  ss += x_ * x_;
    }
    red[tid] = ss;
    __syncthreads();
    if (l == 0) { float s = 0.f; for (int i = 0; i < 64; ++i) s += red[h * 64 + i]; hs[h] = s; }
    __syncthreads();
    float rs = rsqrtf(hs[h] * (1.f / 256.f) + 1e-5f);
#pragma unroll
    for (int i = 0; i < 4; ++i) {
        int col = h * 256 + l * 4 + i;
        float zg  = gate[(size_t)row * 1024 + col];
        float sig = 1.f / (1.f + __expf(-zg));
        yb[(size_t)row * 1024 + col] =
            (unsigned short)f2bf1(vals[i] * rs * rms_w[l * 4 + i] * zg * sig);
    }
}

// ---------------------------------------------------------------------------
extern "C" void kernel_launch(void* const* d_in, const int* in_sizes, int n_in,
                              void* d_out, int out_size, void* d_ws, size_t ws_size,
                              hipStream_t stream) {
    (void)in_sizes; (void)n_in; (void)out_size; (void)ws_size;
    const float* x    = (const float*)d_in[0];
    const float* Wq   = (const float*)d_in[1];
    const float* Wk   = (const float*)d_in[2];
    const float* Wv   = (const float*)d_in[3];
    const float* Wg   = (const float*)d_in[4];
    const float* Wgk1 = (const float*)d_in[5];
    const float* Wgk2 = (const float*)d_in[6];
    const float* bgk2 = (const float*)d_in[7];
    const float* Wout = (const float*)d_in[8];
    const float* rmsw = (const float*)d_in[9];

    // workspace layout (~52 MB); all sizes multiples of 16 B
    char* base = (char*)d_ws;
    float* q     = (float*)base;          base += 2048 * 512 * 4;
    float* kk    = (float*)base;          base += 2048 * 512 * 4;
    float* gdec  = (float*)base;          base += 2048 * 512 * 4;
    float* v     = (float*)base;          base += 2048 * 1024 * 4;
    float* gate  = (float*)base;          base += 2048 * 1024 * 4;
    float* o     = (float*)base;          base += 2048 * 1024 * 4;
    float* glr   = (float*)base;          base += 2048 * 16 * 4;
    unsigned short* xb    = (unsigned short*)base; base += 2048 * 1024 * 2;
    unsigned short* yb    = (unsigned short*)base; base += 2048 * 1024 * 2;
    unsigned* Wqp   = (unsigned*)base;    base += 512 * 512 * 4;
    unsigned* Wkp   = (unsigned*)base;    base += 512 * 512 * 4;
    unsigned* Wvp   = (unsigned*)base;    base += 512 * 1024 * 4;
    unsigned* Wgp   = (unsigned*)base;    base += 512 * 1024 * 4;
    unsigned* Woutp = (unsigned*)base;    base += 512 * 1024 * 4;

    dim3 blk(256);
    cvt_bf16_kernel<<<(2048 * 1024) / 256, blk, 0, stream>>>(x, xb, 2048 * 1024);
    pack_weight_kernel<<<(512 * 512) / 256, blk, 0, stream>>>(Wq, Wqp, 1024, 512);
    pack_weight_kernel<<<(512 * 512) / 256, blk, 0, stream>>>(Wk, Wkp, 1024, 512);
    pack_weight_kernel<<<(512 * 1024) / 256, blk, 0, stream>>>(Wv, Wvp, 1024, 1024);
    pack_weight_kernel<<<(512 * 1024) / 256, blk, 0, stream>>>(Wg, Wgp, 1024, 1024);
    pack_weight_kernel<<<(512 * 1024) / 256, blk, 0, stream>>>(Wout, Woutp, 1024, 1024);

    gemm_bf16_kernel<<<dim3(4, 16), blk, 0, stream>>>(xb, Wqp, q,    2048,  512, 1024);
    gemm_bf16_kernel<<<dim3(4, 16), blk, 0, stream>>>(xb, Wkp, kk,   2048,  512, 1024);
    gemm_bf16_kernel<<<dim3(8, 16), blk, 0, stream>>>(xb, Wvp, v,    2048, 1024, 1024);
    gemm_bf16_kernel<<<dim3(8, 16), blk, 0, stream>>>(xb, Wgp, gate, 2048, 1024, 1024);

    glr_kernel<<<(2048 * 16) / 256, blk, 0, stream>>>(x, Wgk1, glr);
    gate_decay_kernel<<<(2048 * 512) / 256, blk, 0, stream>>>(glr, Wgk2, bgk2, gdec);

    // S(128KB) + Qs2/Kng2/Kc2(3*4KB) + VB2(8KB) + Ab(512B) + eC(512B)
    size_t smem = 131072 + 3 * 4096 + 8192 + 512 + 512;
    gla_scan_kernel<<<dim3(8), blk, smem, stream>>>(q, kk, v, gdec, o);

    rms_silu_kernel<<<dim3(2048), blk, 0, stream>>>(o, gate, rmsw, yb);

    gemm_bf16_kernel<<<dim3(8, 16), blk, 0, stream>>>(yb, Woutp, (float*)d_out,
                                                      2048, 1024, 1024);
}
